// HermesMessageLayer_13005160972931
// MI455X (gfx1250) — matrix operand order repaired and checked by reading, not compiled, not run on hardware
//
#include <hip/hip_runtime.h>
#include <hip/hip_bf16.h>

// out[e,i,n] = einsum('ejm,efr,bfnm,brij->ein') + bias
// Strategy:
//   W2[k=(f,r,j,m)][c=(i*3+n)] = sum_b kernel[b,f,n,m]*weight[b,r,i,j]  (960x96 bf16, in d_ws,
//       pre-swizzled into v_wmma_f32_16x16x32_bf16 B-fragment layout)
//   out[e,c] = sum_k (precomp[e, k/96] * inp[e, k%96]) * W2[k][c] + bias[c/3]
// GEMM M=E=300000, K=960, N=96 via bf16 WMMA with f32 accumulate.
// Each wave computes TWO 16-row tiles (32 edges) so every B-fragment load
// feeds two WMMAs (halves L2 traffic, doubles latency-hiding work per load).

typedef __attribute__((ext_vector_type(16))) __bf16 v16bf;
typedef __attribute__((ext_vector_type(8)))  float  v8f;

#define KDIM 960      // F*R*J*M = 10*96
#define NDIM 96       // I*N = 32*3
#define NKB  30       // KDIM/32
#define NNT  6        // NDIM/16
// fragment: (kb,nt) -> 512 bf16 (1024 B): lane-major, 16 bf16 per lane.
#define W2_ELEMS (NKB * NNT * 32 * 16)   // 92160 == 960*96

// ---------------- Kernel 0: build swizzled bf16 B fragments of W2 -------------
__global__ void build_w2_frags(const float* __restrict__ kern,   // [B=6,F=5,N=3,M=3]
                               const float* __restrict__ weight, // [B=6,R=2,I=32,J=32]
                               __bf16* __restrict__ wsB) {
  int tid = blockIdx.x * blockDim.x + threadIdx.x;
  if (tid >= W2_ELEMS) return;
  int e16     = tid & 15;          // element within lane's 16 bf16
  int lane    = (tid >> 4) & 31;   // lane within fragment
  int fragIdx = tid >> 9;          // (kb*6 + nt)
  int kb = fragIdx / NNT;
  int nt = fragIdx % NNT;
  int g  = e16 >> 1;               // VGPR index 0..7
  int p  = e16 & 1;                // packed low/high
  int kh = lane >> 4;              // K half
  int nc = lane & 15;              // column within 16-wide N tile
  // B k order chosen here (same as A): lane(kh,nc), VGPR g holds K = 16*kh + 2g + {0,1}
  int k = kb * 32 + kh * 16 + g * 2 + p;   // 0..959
  int c = nt * 16 + nc;                    // 0..95  (c = i*3 + n)
  int fr = k / 96, jm = k % 96;
  int f = fr >> 1, r = fr & 1;
  int j = jm / 3,  m = jm % 3;
  int i = c / 3,   n = c % 3;
  float acc = 0.f;
  #pragma unroll
  for (int b = 0; b < 6; ++b)
    acc += kern[b * 45 + f * 9 + n * 3 + m] * weight[b * 2048 + r * 1024 + i * 32 + j];
  wsB[tid] = (__bf16)acc;
}

// ---------------- Kernel 1: main WMMA GEMM ------------------------------------
// 128 threads = 4 waves; each wave computes a 32(edges) x 96 output tile
// (two 16-row WMMA tiles sharing every B fragment).
__global__ void __launch_bounds__(128)
hermes_wmma_gemm(const float* __restrict__ inp,      // [E, 96]
                 const float* __restrict__ precomp,  // [E, 10]
                 const __bf16* __restrict__ wsB,     // swizzled W2 fragments
                 const float* __restrict__ bias,     // [32]
                 float* __restrict__ out,            // [E, 96]
                 int E) {
  __shared__ float sX[4 * 32 * 96];                  // 48 KB: per-wave fp32 inp rows

  const int w    = threadIdx.x >> 5;
  const int lane = threadIdx.x & 31;
  const int row  = lane & 15;                        // matrix row within a 16-row tile
  const int kh   = lane >> 4;                        // K-half / C-row-half selector
  const long e0  = (long)(blockIdx.x * 4 + w) * 32;  // wave's first edge
  const bool act0 = (e0 + 16 <= (long)E);            // tile 0 fully valid
  const bool act1 = (e0 + 32 <= (long)E);            // tile 1 fully valid

  // Stage this wave's 32x96 fp32 inp rows into LDS (coalesced 512B/iter).
  {
    const float4* src = (const float4*)(inp + e0 * 96);
    float4* dst = (float4*)(&sX[w * 3072]);
    if (act1) {
      #pragma unroll
      for (int it = 0; it < 24; ++it)
        dst[it * 32 + lane] = src[it * 32 + lane];
    } else if (act0) {
      #pragma unroll
      for (int it = 0; it < 12; ++it)
        dst[it * 32 + lane] = src[it * 32 + lane];
    }
  }
  __syncthreads();   // all threads reach this (no early returns)

  if (act0) {
    v8f acc0[NNT], acc1[NNT];
    #pragma unroll
    for (int nt = 0; nt < NNT; ++nt) {
      acc0[nt] = (v8f){0.f,0.f,0.f,0.f,0.f,0.f,0.f,0.f};
      acc1[nt] = (v8f){0.f,0.f,0.f,0.f,0.f,0.f,0.f,0.f};
    }

    const float* sRow0 = &sX[w * 3072 + row * 96];          // tile 0 row
    const float* sRow1 = &sX[w * 3072 + 1536 + row * 96];   // tile 1 row
    const char*  wsBc  = (const char*)wsB;

    for (int fr = 0; fr < 10; ++fr) {
      const float p0 = precomp[(e0 + row) * 10 + fr];
      const float p1 = act1 ? precomp[(e0 + 16 + row) * 10 + fr] : 0.f;
      #pragma unroll
      for (int kc = 0; kc < 3; ++kc) {
        const int kb  = fr * 3 + kc;
        const int jmb = kc * 32;
        // Build both A fragments: Z[row, kb*32+koff] = p * inp[row, jmb+koff], bf16.
        v16bf a0, a1;
        #pragma unroll
        for (int g = 0; g < 8; ++g) {
          const int koff = 2 * g + 16 * kh;
          const int jm = jmb + koff;
          const float x00 = sRow0[jm],     x01 = sRow0[jm + 1];
          const float x10 = sRow1[jm],     x11 = sRow1[jm + 1];
          a0[2 * g]     = (__bf16)(x00 * p0);
          a0[2 * g + 1] = (__bf16)(x01 * p0);
          a1[2 * g]     = (__bf16)(x10 * p1);
          a1[2 * g + 1] = (__bf16)(x11 * p1);
        }
        // Each B fragment feeds two WMMAs (tile 0 + tile 1).
        #pragma unroll
        for (int nt = 0; nt < NNT; ++nt) {
          const v16bf b = *(const v16bf*)(wsBc + (((kb * NNT + nt) << 10) + (lane << 5)));
          acc0[nt] = __builtin_amdgcn_wmma_f32_16x16x32_bf16(
              false, a0, false, b, (short)0, acc0[nt], false, false);
          acc1[nt] = __builtin_amdgcn_wmma_f32_16x16x32_bf16(
              false, a1, false, b, (short)0, acc1[nt], false, false);
        }
      }
    }

    // Epilogue: D layout -> lane holds col (lane&15), rows v + 8*kh for v=0..7.
    #pragma unroll
    for (int nt = 0; nt < NNT; ++nt) {
      const int c  = nt * 16 + (lane & 15);
      const float bi = bias[c / 3];
      #pragma unroll
      for (int v = 0; v < 8; ++v) {
        const long rowe = e0 + v + 8 * kh;
        out[rowe * 96 + c] = acc0[nt][v] + bi;
        if (act1) out[(rowe + 16) * 96 + c] = acc1[nt][v] + bi;
      }
    }
  }
}

// ---------------- Launch ------------------------------------------------------
extern "C" void kernel_launch(void* const* d_in, const int* in_sizes, int n_in,
                              void* d_out, int out_size, void* d_ws, size_t ws_size,
                              hipStream_t stream) {
  const float* inp     = (const float*)d_in[0];   // [E,32,3]
  const float* precomp = (const float*)d_in[1];   // [E,5,2]
  const float* kern    = (const float*)d_in[2];   // [6,5,3,3]
  const float* weight  = (const float*)d_in[3];   // [6,2,32,32]
  const float* bias    = (const float*)d_in[4];   // [32]
  float* out = (float*)d_out;
  __bf16* wsB = (__bf16*)d_ws;                    // needs 184320 bytes

  const int E = in_sizes[0] / 96;

  build_w2_frags<<<(W2_ELEMS + 255) / 256, 256, 0, stream>>>(kern, weight, wsB);

  const int waveJobs = (E + 31) / 32;             // 32 edges per wave
  const int blocks   = (waveJobs + 3) / 4;        // 4 waves per block
  hermes_wmma_gemm<<<blocks, 128, 0, stream>>>(inp, precomp, wsB, bias, out, E);
}
